// MultiLayerTransformerNeck_29807073034405
// MI455X (gfx1250) — matrix operand order repaired and checked by reading, not compiled, not run on hardware
//
#include <hip/hip_runtime.h>
#include <hip/hip_bf16.h>
#include <math.h>

// ---------------------------------------------------------------------------
// Types for CDNA5 WMMA (wave32, 16x16x32 f16 -> f32)
// ---------------------------------------------------------------------------
typedef __attribute__((ext_vector_type(16))) _Float16 v16h;
typedef __attribute__((ext_vector_type(8)))  _Float16 v8h;
typedef __attribute__((ext_vector_type(4)))  _Float16 v4h;
typedef __attribute__((ext_vector_type(8)))  float    v8f;

#define FLAG_ACC  1
#define FLAG_GELU 2

static inline unsigned ceil_div_h(long a, long b) { return (unsigned)((a + b - 1) / b); }

__device__ __forceinline__ int imin(int a, int b) { return a < b ? a : b; }

// ---------------------------------------------------------------------------
// Generic batched GEMM:  C[z] = epilogue( alpha * A[z] * op(B[z]) + bias )
//   A: [M,K] row-major (lda).  B: transB ? [N,K] : [K,N] (ldb).  C: [M,N] (ldc)
//   batch z = zo*innerZ + zi with separate outer/inner strides (window x head).
// 64x64 block tile, 128 threads = 4 waves, each wave a 32x32 quadrant as
// 2x2 v_wmma_f32_16x16x32_f16 accumulators. K staged through LDS in f16.
//
// M/N edges are handled by CLAMPING load indices (out-of-range rows/cols only
// feed outputs the epilogue masks), so all full-K tiles use unguarded
// vectorized float4 loads. Only a K-remainder tile (K%32!=0) takes the
// guarded scalar path with zero fill.
// ---------------------------------------------------------------------------
__global__ __launch_bounds__(128)
void gemm_wmma_kernel(const float* __restrict__ A, long lda,
                      const float* __restrict__ B, long ldb, int transB,
                      const float* __restrict__ bias,
                      float* __restrict__ C, long ldc,
                      int M, int N, int K, float alpha, int flags, int innerZ,
                      long sAo, long sAi, long sBo, long sBi, long sCo, long sCi)
{
    __shared__ __align__(16) _Float16 sA[64 * 40];  // [m][k], row stride 80B
    __shared__ __align__(16) _Float16 sB[64 * 40];  // [n][k]

    const int z  = blockIdx.z;
    const int zo = z / innerZ;
    const int zi = z - zo * innerZ;
    A += (long)zo * sAo + (long)zi * sAi;
    B += (long)zo * sBo + (long)zi * sBi;
    C += (long)zo * sCo + (long)zi * sCi;

    const int tid    = threadIdx.x;
    const int lane   = tid & 31;
    const int wave   = tid >> 5;
    const int rowBlk = blockIdx.y * 64;
    const int colBlk = blockIdx.x * 64;
    const int wm     = (wave >> 1) * 32;
    const int wn     = (wave & 1) * 32;

    v8f acc[2][2];
#pragma unroll
    for (int a = 0; a < 2; ++a)
#pragma unroll
        for (int b = 0; b < 2; ++b)
#pragma unroll
            for (int e = 0; e < 8; ++e) acc[a][b][e] = 0.0f;

    // Fragment offsets per the ISA 16-bit A / B VGPR layouts (wave32).
    const int mrow = lane & 15;
    const int aOff = (lane < 16) ? 0 : 8;    // A: K {0..7,16..23} vs {8..15,24..31}
    const int bOff = (lane < 16) ? 0 : 8;    // B: K {0..7,16..23} vs {8..15,24..31}

    // Fast-path thread mapping: 64x32 tile = 16 rows/iter of 8 float4s.
    const int fa = tid & 7;                  // float4 slot within 32-wide row
    const int ra = tid >> 3;                 // row 0..15 (stride 16 over 4 iters)
    const int nb = tid & 15;                 // NN: float4 slot within 64-wide n row
    const int kb = tid >> 4;                 // NN: k row 0..7 (stride 8 over 4 iters)

    const int kFull = K & ~31;

    for (int k0 = 0; k0 < kFull; k0 += 32) {
        // Speculative prefetch of the next K tile (dropped if invalid).
        {
            int pr = imin(rowBlk + ra, M - 1);
            __builtin_prefetch(&A[(long)pr * lda + k0 + 32], 0, 1);
        }
        // ---- A tile: unguarded vector loads, rows clamped ----
#pragma unroll
        for (int ri = 0; ri < 4; ++ri) {
            int row = ra + ri * 16;
            int gr  = imin(rowBlk + row, M - 1);
            const float4 v = *(const float4*)&A[(long)gr * lda + k0 + fa * 4];
            v4h h; h.x = (_Float16)v.x; h.y = (_Float16)v.y;
                   h.z = (_Float16)v.z; h.w = (_Float16)v.w;
            *(v4h*)&sA[row * 40 + fa * 4] = h;
        }
        // ---- B tile ----
        if (transB) {
#pragma unroll
            for (int ri = 0; ri < 4; ++ri) {
                int row = ra + ri * 16;
                int gn  = imin(colBlk + row, N - 1);
                const float4 v = *(const float4*)&B[(long)gn * ldb + k0 + fa * 4];
                v4h h; h.x = (_Float16)v.x; h.y = (_Float16)v.y;
                       h.z = (_Float16)v.z; h.w = (_Float16)v.w;
                *(v4h*)&sB[row * 40 + fa * 4] = h;
            }
        } else {
            int nAddr = imin(colBlk + nb * 4, (N - 4) > 0 ? (N - 4) : 0);
#pragma unroll
            for (int ki = 0; ki < 4; ++ki) {
                int k = kb + ki * 8;
                const float4 v = *(const float4*)&B[(long)(k0 + k) * ldb + nAddr];
                sB[(nb * 4 + 0) * 40 + k] = (_Float16)v.x;
                sB[(nb * 4 + 1) * 40 + k] = (_Float16)v.y;
                sB[(nb * 4 + 2) * 40 + k] = (_Float16)v.z;
                sB[(nb * 4 + 3) * 40 + k] = (_Float16)v.w;
            }
        }
        __syncthreads();
#pragma unroll
        for (int tm = 0; tm < 2; ++tm) {
            const _Float16* ap = &sA[(wm + tm * 16 + mrow) * 40 + aOff];
            v8h alo = *(const v8h*)(ap);
            v8h ahi = *(const v8h*)(ap + 16);
            v16h afrag;
#pragma unroll
            for (int q = 0; q < 8; ++q) { afrag[q] = alo[q]; afrag[q + 8] = ahi[q]; }
#pragma unroll
            for (int tn = 0; tn < 2; ++tn) {
                const _Float16* bp = &sB[(wn + tn * 16 + mrow) * 40 + bOff];
                v8h blo = *(const v8h*)(bp);
                v8h bhi = *(const v8h*)(bp + 16);
                v16h bfrag;
#pragma unroll
                for (int q = 0; q < 8; ++q) { bfrag[q] = blo[q]; bfrag[q + 8] = bhi[q]; }
                acc[tm][tn] = __builtin_amdgcn_wmma_f32_16x16x32_f16(
                    false, afrag, false, bfrag, (short)0, acc[tm][tn], false, false);
            }
        }
        __syncthreads();
    }

    // ---- K remainder tile (K % 32 != 0): guarded scalar with zero fill ----
    if (kFull < K) {
        const int k0 = kFull;
#pragma unroll
        for (int it = 0; it < 16; ++it) {
            int e = tid + it * 128;
            int r = e >> 5, c = e & 31;
            int gk = k0 + c;
            int gr = imin(rowBlk + r, M - 1);
            int gn = imin(colBlk + r, N - 1);
            float va = (gk < K) ? A[(long)gr * lda + gk] : 0.0f;
            float vb = 0.0f;
            if (gk < K) vb = transB ? B[(long)gn * ldb + gk] : B[(long)gk * ldb + gn];
            sA[r * 40 + c] = (_Float16)va;
            sB[r * 40 + c] = (_Float16)vb;
        }
        __syncthreads();
#pragma unroll
        for (int tm = 0; tm < 2; ++tm) {
            const _Float16* ap = &sA[(wm + tm * 16 + mrow) * 40 + aOff];
            v8h alo = *(const v8h*)(ap);
            v8h ahi = *(const v8h*)(ap + 16);
            v16h afrag;
#pragma unroll
            for (int q = 0; q < 8; ++q) { afrag[q] = alo[q]; afrag[q + 8] = ahi[q]; }
#pragma unroll
            for (int tn = 0; tn < 2; ++tn) {
                const _Float16* bp = &sB[(wn + tn * 16 + mrow) * 40 + bOff];
                v8h blo = *(const v8h*)(bp);
                v8h bhi = *(const v8h*)(bp + 16);
                v16h bfrag;
#pragma unroll
                for (int q = 0; q < 8; ++q) { bfrag[q] = blo[q]; bfrag[q + 8] = bhi[q]; }
                acc[tm][tn] = __builtin_amdgcn_wmma_f32_16x16x32_f16(
                    false, afrag, false, bfrag, (short)0, acc[tm][tn], false, false);
            }
        }
        __syncthreads();
    }

    // ---- Epilogue ----
#pragma unroll
    for (int tm = 0; tm < 2; ++tm) {
#pragma unroll
        for (int tn = 0; tn < 2; ++tn) {
            int baseR = rowBlk + wm + tm * 16 + ((lane < 16) ? 0 : 8);
            int cN    = colBlk + wn + tn * 16 + (lane & 15);
            if (cN < N) {
                float bv = bias ? bias[cN] : 0.0f;
#pragma unroll
                for (int r = 0; r < 8; ++r) {
                    int gr = baseR + r;
                    if (gr < M) {
                        float v = acc[tm][tn][r] * alpha + bv;
                        if (flags & FLAG_GELU) v = 0.5f * v * (1.0f + erff(v * 0.70710678118654752f));
                        long o = (long)gr * ldc + cN;
                        if (flags & FLAG_ACC) v += C[o];
                        C[o] = v;
                    }
                }
            }
        }
    }
}

// ---------------------------------------------------------------------------
// LayerNorm over last dim (per row), eps = 1e-6, affine g/b.
// ---------------------------------------------------------------------------
__global__ __launch_bounds__(256)
void ln_kernel(const float* __restrict__ in, float* __restrict__ out,
               const float* __restrict__ g, const float* __restrict__ b, int C)
{
    __shared__ float red[256];
    const int row = blockIdx.x;
    const int tid = threadIdx.x;
    const float* x = in + (long)row * C;
    float s = 0.0f;
    for (int c = tid; c < C; c += 256) s += x[c];
    red[tid] = s; __syncthreads();
    for (int st = 128; st > 0; st >>= 1) { if (tid < st) red[tid] += red[tid + st]; __syncthreads(); }
    float mean = red[0] / (float)C; __syncthreads();
    float s2 = 0.0f;
    for (int c = tid; c < C; c += 256) { float d = x[c] - mean; s2 += d * d; }
    red[tid] = s2; __syncthreads();
    for (int st = 128; st > 0; st >>= 1) { if (tid < st) red[tid] += red[tid + st]; __syncthreads(); }
    float inv = rsqrtf(red[0] / (float)C + 1e-6f); __syncthreads();
    float* y = out + (long)row * C;
    for (int c = tid; c < C; c += 256) y[c] = (x[c] - mean) * inv * g[c] + b[c];
}

// ---------------------------------------------------------------------------
// Window partition (zero pad) and reverse (+residual add). x: [B,H,H,C] BHWC.
// ---------------------------------------------------------------------------
__global__ __launch_bounds__(256)
void winpart_kernel(const float* __restrict__ x, float* __restrict__ xw,
                    int Bn, int H, int win, int nW, int C, long total)
{
    long idx = (long)blockIdx.x * 256 + threadIdx.x;
    if (idx >= total) return;
    int  c = (int)(idx % C);  long t = idx / C;
    int  j = (int)(t % win);  t /= win;
    int  i = (int)(t % win);  t /= win;
    int ww = (int)(t % nW);   t /= nW;
    int wh = (int)(t % nW);   int b = (int)(t / nW);
    int h = wh * win + i, w = ww * win + j;
    xw[idx] = (h < H && w < H) ? x[(((long)b * H + h) * H + w) * C + c] : 0.0f;
}

__global__ __launch_bounds__(256)
void winunpart_add_kernel(const float* __restrict__ t, float* __restrict__ x,
                          int Bn, int H, int win, int nW, int C, long total)
{
    long idx = (long)blockIdx.x * 256 + threadIdx.x;
    if (idx >= total) return;
    int c = (int)(idx % C); long q = idx / C;
    int w = (int)(q % H);   q /= H;
    int h = (int)(q % H);   int b = (int)(q / H);
    int wh = h / win, i = h % win, ww = w / win, j = w % win;
    long row = ((((long)b * nW + wh) * nW + ww) * win + i) * win + j;
    x[idx] += t[row * C + c];
}

// ---------------------------------------------------------------------------
// Decomposed relative-position bias: reh/rew dots, then add to scores.
// ---------------------------------------------------------------------------
__global__ __launch_bounds__(256)
void relpos_kernel(const float* __restrict__ qkv, const float* __restrict__ relh,
                   const float* __restrict__ relw, float* __restrict__ rel,
                   int win, int L, int heads, long total)
{
    long idx = (long)blockIdx.x * 256 + threadIdx.x;
    if (idx >= total) return;
    int k = (int)(idx % win); long t = idx / win;
    int which = (int)(t & 1); t >>= 1;
    int q = (int)(t % L);     int z = (int)(t / L);
    int w = z / heads, n = z - w * heads;
    int qh = q / win, qw = q - qh * win;
    const float* qp = qkv + ((long)w * L + q) * 2304 + (long)n * 48;
    int pos = ((which == 0) ? qh : qw) - k + win - 1;
    const float* rp = ((which == 0) ? relh : relw) + (long)pos * 48;
    float s = 0.0f;
    for (int d = 0; d < 48; ++d) s += qp[d] * rp[d];
    rel[idx] = s;
}

__global__ __launch_bounds__(256)
void add_relpos_kernel(float* __restrict__ sc, const float* __restrict__ rel,
                       int win, int L, long total)
{
    long idx = (long)blockIdx.x * 256 + threadIdx.x;
    if (idx >= total) return;
    int ki = (int)(idx % L); long t = idx / L;
    int q  = (int)(t % L);   long z = t / L;
    int kh = ki / win, kw = ki - kh * win;
    const float* r = rel + ((z * L + q) * 2) * (long)win;
    sc[idx] += r[kh] + r[win + kw];
}

// ---------------------------------------------------------------------------
// Row softmax (stable), row length L.
// ---------------------------------------------------------------------------
__global__ __launch_bounds__(256)
void softmax_kernel(float* __restrict__ s, int L)
{
    __shared__ float red[256];
    const int tid = threadIdx.x;
    float* p = s + (long)blockIdx.x * L;
    float m = -1e30f;
    for (int i = tid; i < L; i += 256) m = fmaxf(m, p[i]);
    red[tid] = m; __syncthreads();
    for (int st = 128; st > 0; st >>= 1) { if (tid < st) red[tid] = fmaxf(red[tid], red[tid + st]); __syncthreads(); }
    m = red[0]; __syncthreads();
    float sum = 0.0f;
    for (int i = tid; i < L; i += 256) { float e = expf(p[i] - m); p[i] = e; sum += e; }
    red[tid] = sum; __syncthreads();
    for (int st = 128; st > 0; st >>= 1) { if (tid < st) red[tid] += red[tid + st]; __syncthreads(); }
    float inv = 1.0f / red[0]; __syncthreads();
    for (int i = tid; i < L; i += 256) p[i] *= inv;
}

// ---------------------------------------------------------------------------
// im2col: NCHW source (patch embeds) and BHWC source (neck 3x3, pad=1).
// Column order (c,kh,kw) matches OIHW weight flattening.
// ---------------------------------------------------------------------------
__global__ __launch_bounds__(256)
void im2col_chw_kernel(const float* __restrict__ src, float* __restrict__ dst,
                       int Cin, int H, int W, int KH, int KW, int stride,
                       int OH, int OW, long total)
{
    long idx = (long)blockIdx.x * 256 + threadIdx.x;
    if (idx >= total) return;
    long Kc = (long)Cin * KH * KW;
    int col = (int)(idx % Kc); long row = idx / Kc;
    int ow = (int)(row % OW); long t = row / OW;
    int oh = (int)(t % OH);   int b = (int)(t / OH);
    int kw = col % KW; int kh = (col / KW) % KH; int c = col / (KW * KH);
    int ih = oh * stride + kh, iw = ow * stride + kw;
    dst[idx] = src[(((long)b * Cin + c) * H + ih) * W + iw];
}

__global__ __launch_bounds__(256)
void im2col_hwc3_kernel(const float* __restrict__ src, float* __restrict__ dst,
                        int Cin, int H, int W, long total)
{
    long idx = (long)blockIdx.x * 256 + threadIdx.x;
    if (idx >= total) return;
    long Kc = (long)Cin * 9;
    int col = (int)(idx % Kc); long row = idx / Kc;
    int ow = (int)(row % W); long t = row / W;
    int oh = (int)(t % H);   int b = (int)(t / H);
    int kw = col % 3; int kh = (col / 3) % 3; int c = col / 9;
    int ih = oh + kh - 1, iw = ow + kw - 1;
    float v = 0.0f;
    if (ih >= 0 && ih < H && iw >= 0 && iw < W)
        v = src[(((long)b * H + ih) * W + iw) * Cin + c];
    dst[idx] = v;
}

// ---------------------------------------------------------------------------
// Deconv2 scatter: [B*16*16, 3072=(o,k,l)] -> BHWC 32x32x768 (+= with bias).
// ---------------------------------------------------------------------------
__global__ __launch_bounds__(256)
void deconv_scatter_kernel(const float* __restrict__ dt, const float* __restrict__ bias,
                           float* __restrict__ x, long total)
{
    long idx = (long)blockIdx.x * 256 + threadIdx.x;
    if (idx >= total) return;
    int o = (int)(idx % 768); long t = idx / 768;
    int Wo = (int)(t % 32); t /= 32;
    int Ho = (int)(t % 32); int b = (int)(t / 32);
    int h = Ho >> 1, kk = Ho & 1, w = Wo >> 1, ll = Wo & 1;
    x[idx] += dt[(((long)b * 16 + h) * 16 + w) * 3072 + (long)o * 4 + kk * 2 + ll] + bias[o];
}

__global__ __launch_bounds__(256)
void finalize_embed_kernel(float* __restrict__ x, const float* __restrict__ level,
                           const float* __restrict__ pos, long total)
{
    long idx = (long)blockIdx.x * 256 + threadIdx.x;
    if (idx >= total) return;
    int c = (int)(idx % 768);
    int tkn = (int)((idx / 768) % 1024);
    x[idx] += level[c] + level[768 + c] + level[1536 + c] + level[2304 + c]
            + pos[(long)tkn * 768 + c];
}

__global__ __launch_bounds__(256)
void hwc2chw_kernel(const float* __restrict__ in, float* __restrict__ out, long total)
{
    long idx = (long)blockIdx.x * 256 + threadIdx.x;
    if (idx >= total) return;
    int w = (int)(idx % 32); long t = idx / 32;
    int h = (int)(t % 32); t /= 32;
    int c = (int)(t % 256); int b = (int)(t / 256);
    out[idx] = in[(((long)b * 32 + h) * 32 + w) * 256 + c];
}

// ---------------------------------------------------------------------------
// Host orchestration
// ---------------------------------------------------------------------------
static void gemm(hipStream_t st, const float* A, long lda, const float* B, long ldb,
                 int transB, const float* bias, float* C, long ldc,
                 int M, int N, int K, float alpha, int flags,
                 int Z = 1, int inner = 1,
                 long sAo = 0, long sAi = 0, long sBo = 0, long sBi = 0,
                 long sCo = 0, long sCi = 0)
{
    dim3 g((unsigned)((N + 63) / 64), (unsigned)((M + 63) / 64), (unsigned)Z);
    hipLaunchKernelGGL(gemm_wmma_kernel, g, dim3(128), 0, st,
                       A, lda, B, ldb, transB, bias, C, ldc, M, N, K,
                       alpha, flags, inner, sAo, sAi, sBo, sBi, sCo, sCi);
}

#define EW(kernel, total, ...)                                                        \
    hipLaunchKernelGGL(kernel, dim3(ceil_div_h((total), 256)), dim3(256), 0, stream,  \
                       __VA_ARGS__)

extern "C" void kernel_launch(void* const* d_in, const int* in_sizes, int n_in,
                              void* d_out, int out_size, void* d_ws, size_t ws_size,
                              hipStream_t stream)
{
    (void)in_sizes; (void)n_in; (void)out_size; (void)ws_size;

    const float* feat0 = (const float*)d_in[0];   // (2,256,128,128)
    const float* feat1 = (const float*)d_in[1];   // (2,512,64,64)
    const float* feat2 = (const float*)d_in[2];   // (2,1024,32,32)
    const float* feat3 = (const float*)d_in[3];   // (2,2048,16,16)

    // jax pytree flatten order: blocks(5 x 14 sorted-key leaves), level, neck(6), patch(4 x 2), pos
    enum { FC1_B = 0, FC1_W, FC2_B, FC2_W, LN1_B, LN1_G, LN2_B, LN2_G,
           PROJ_B, PROJ_W, QKV_B, QKV_W, REL_H, REL_W };
    #define BLKP(i, leaf) ((const float*)d_in[4 + (i) * 14 + (leaf)])
    const float* level   = (const float*)d_in[74];
    const float* nc1_w   = (const float*)d_in[75];
    const float* nc2_w   = (const float*)d_in[76];
    const float* nln1_b  = (const float*)d_in[77];
    const float* nln1_g  = (const float*)d_in[78];
    const float* nln2_b  = (const float*)d_in[79];
    const float* nln2_g  = (const float*)d_in[80];
    const float* pb[4], * pw[4];
    for (int i = 0; i < 4; ++i) { pb[i] = (const float*)d_in[81 + i * 2]; pw[i] = (const float*)d_in[82 + i * 2]; }
    const float* pos = (const float*)d_in[89];

    // Workspace arena
    size_t off = 0;
    auto alloc = [&](size_t nfl) -> float* {
        float* p = (float*)((char*)d_ws + off);
        off += ((nfl * sizeof(float)) + 255) & ~(size_t)255;
        return p;
    };
    float* X    = alloc(2048L * 768);          // residual stream, BHWC tokens
    float* XN   = alloc(2048L * 768);
    float* XW   = alloc(3528L * 768);          // window tokens (max T)
    float* QKV  = alloc(3528L * 2304);
    float* SC   = alloc(32L * 1024 * 1024);    // attention scores (max: global block)
    float* REL  = alloc(32L * 1024 * 32 * 2);
    float* AO   = alloc(3528L * 768);
    float* PRJ  = alloc(3528L * 768);
    float* H1   = alloc(2048L * 3072);
    float* H2   = alloc(2048L * 768);
    float* COL  = alloc(2048L * 4096);         // im2col staging (max: patch0)
    float* N1   = alloc(2048L * 256);
    float* N2   = alloc(2048L * 256);
    float* DT   = alloc(512L * 3072);

    const float rs48 = 0.14433756729740643f;   // 48^-0.5

    // -------- Patch embeds, summed into X (BHWC, 2x32x32x768) --------
    hipMemsetAsync(X, 0, 2048L * 768 * sizeof(float), stream);

    EW(im2col_chw_kernel, 2048L * 4096, feat0, COL, 256, 128, 128, 4, 4, 4, 32, 32, 2048L * 4096);
    gemm(stream, COL, 4096, pw[0], 4096, 1, pb[0], X, 768, 2048, 768, 4096, 1.0f, FLAG_ACC);
    EW(im2col_chw_kernel, 2048L * 2048, feat1, COL, 512, 64, 64, 2, 2, 2, 32, 32, 2048L * 2048);
    gemm(stream, COL, 2048, pw[1], 2048, 1, pb[1], X, 768, 2048, 768, 2048, 1.0f, FLAG_ACC);
    EW(im2col_chw_kernel, 2048L * 1024, feat2, COL, 1024, 32, 32, 1, 1, 1, 32, 32, 2048L * 1024);
    gemm(stream, COL, 1024, pw[2], 1024, 1, pb[2], X, 768, 2048, 768, 1024, 1.0f, FLAG_ACC);
    EW(im2col_chw_kernel, 512L * 2048, feat3, COL, 2048, 16, 16, 1, 1, 1, 16, 16, 512L * 2048);
    gemm(stream, COL, 2048, pw[3], 3072, 0, nullptr, DT, 3072, 512, 3072, 2048, 1.0f, 0);
    EW(deconv_scatter_kernel, 2048L * 768, DT, pb[3], X, 2048L * 768);
    EW(finalize_embed_kernel, 2048L * 768, X, level, pos, 2048L * 768);

    // -------- Transformer blocks --------
    for (int i = 0; i < 5; ++i) {
        const int win  = (i < 4) ? 14 : 32;      // global block == one 32x32 window
        const int nW   = (win == 14) ? 3 : 1;
        const int L    = win * win;              // 196 or 1024
        const int nWin = 2 * nW * nW;            // 18 or 2
        const int T    = nWin * L;               // 3528 or 2048
        const int Z    = nWin * 16;              // 288 or 32

        hipLaunchKernelGGL(ln_kernel, dim3(2048), dim3(256), 0, stream,
                           X, XN, BLKP(i, LN1_G), BLKP(i, LN1_B), 768);
        EW(winpart_kernel, (long)T * 768, XN, XW, 2, 32, win, nW, 768, (long)T * 768);

        gemm(stream, XW, 768, BLKP(i, QKV_W), 768, 1, BLKP(i, QKV_B), QKV, 2304,
             T, 2304, 768, 1.0f, 0);

        EW(relpos_kernel, (long)Z * L * win * 2, QKV, BLKP(i, REL_H), BLKP(i, REL_W),
           REL, win, L, 16, (long)Z * L * win * 2);

        gemm(stream, QKV, 2304, QKV + 768, 2304, 1, nullptr, SC, L,
             L, L, 48, rs48, 0, Z, 16,
             (long)L * 2304, 48, (long)L * 2304, 48, 16L * L * L, (long)L * L);
        EW(add_relpos_kernel, (long)Z * L * L, SC, REL, win, L, (long)Z * L * L);
        hipLaunchKernelGGL(softmax_kernel, dim3((unsigned)((long)Z * L)), dim3(256), 0, stream, SC, L);

        gemm(stream, SC, L, QKV + 1536, 2304, 0, nullptr, AO, 768,
             L, 48, L, 1.0f, 0, Z, 16,
             16L * L * L, (long)L * L, (long)L * 2304, 48, (long)L * 768, 48);

        gemm(stream, AO, 768, BLKP(i, PROJ_W), 768, 1, BLKP(i, PROJ_B), PRJ, 768,
             T, 768, 768, 1.0f, 0);
        EW(winunpart_add_kernel, 2048L * 768, PRJ, X, 2, 32, win, nW, 768, 2048L * 768);

        hipLaunchKernelGGL(ln_kernel, dim3(2048), dim3(256), 0, stream,
                           X, H2, BLKP(i, LN2_G), BLKP(i, LN2_B), 768);
        gemm(stream, H2, 768, BLKP(i, FC1_W), 768, 1, BLKP(i, FC1_B), H1, 3072,
             2048, 3072, 768, 1.0f, FLAG_GELU);
        gemm(stream, H1, 3072, BLKP(i, FC2_W), 3072, 1, BLKP(i, FC2_B), X, 768,
             2048, 768, 3072, 1.0f, FLAG_ACC);
    }

    // -------- Neck --------
    gemm(stream, X, 768, nc1_w, 768, 1, nullptr, N1, 256, 2048, 256, 768, 1.0f, 0);
    hipLaunchKernelGGL(ln_kernel, dim3(2048), dim3(256), 0, stream, N1, N2, nln1_g, nln1_b, 256);
    EW(im2col_hwc3_kernel, 2048L * 2304, N2, COL, 256, 32, 32, 2048L * 2304);
    gemm(stream, COL, 2304, nc2_w, 2304, 1, nullptr, N1, 256, 2048, 256, 2304, 1.0f, 0);
    hipLaunchKernelGGL(ln_kernel, dim3(2048), dim3(256), 0, stream, N1, N2, nln2_g, nln2_b, 256);
    EW(hwc2chw_kernel, 2L * 256 * 1024, N2, (float*)d_out, 2L * 256 * 1024);
}
